// AttentionTF_80711025427249
// MI455X (gfx1250) — matrix operand order repaired and checked by reading, not compiled, not run on hardware
//
#include <hip/hip_runtime.h>
#include <cstdint>
#include <cstddef>

// ---------------------------------------------------------------------------
// MI455X (gfx1250, wave32) attention pipeline in bf16 WMMA with f32 accumulate.
//   B=4, D=1024, N=2048, K=1024
// All B-operands live in a "K16-grouped" layout: element (k, n) stored at
//   [ ((k/16)*Ncols + n)*16 + (k%16) ]
// so a WMMA B-fragment (lane l/half h holds K = h*16 + 2v, 2v+1 at column l)
// is exactly 32 contiguous bytes per lane -> two global_load_b128.
// ---------------------------------------------------------------------------

typedef __attribute__((ext_vector_type(16))) __bf16   bf16x16;
typedef __attribute__((ext_vector_type(8)))  float    f32x8;
typedef __attribute__((ext_vector_type(4)))  uint32_t u32x4;

union ABFrag { uint32_t u[8]; u32x4 q[2]; bf16x16 v; };

static constexpr int CB = 4;     // batch
static constexpr int CD = 1024;  // feature dim
static constexpr int CN = 2048;  // sequence
static constexpr int CK = 1024;  // projection dim

// ---------------------------------------------------------------------------
// bf16 WMMA GEMM: C(M x Nc) = A(M x Kt, row-major bf16) * B(Kt x Nc, K16-grouped
// bf16).  128 threads = 4 waves; workgroup tile 64x128, wave tile 32x64
// (2x4 WMMA 16x16 tiles -> 8 accumulating WMMAs per k-step of 32).
//
// Fragment layouts used here (wave32; A and B share contiguous k order):
//   A 16x32:  lane l(0..15)/half h: m=l, VGPR v holds K = h*16 + 2v (+1)
//             -> two 16B loads at A[row][k0 + h*16] and A[row][k0 + h*16 + 8]
//   B 32x16:  lane l/half h: n=l, VGPR v holds K = h*16 + 2v (+1)
//             -> two 16B loads from the K16-grouped buffer
//   C 16x16 f32: VGPR v, lane l/half h: m = v + 8h, n = l
//
// EPI: 0 = store f32 row-major          (logits S)
//      1 = store bf16 K16-grouped       (feeds next GEMM's B operand)
//      2 = store bf16 transposed        (Q^T, row-major N x M)
//      3 = residual + f32 row-major     (final output)
// ---------------------------------------------------------------------------
template <int EPI, bool CAUSAL_SKIP, bool KLIMIT>
__global__ __launch_bounds__(128)
void gemm_bf16_wmma(const __bf16* __restrict__ A, const __bf16* __restrict__ Bp,
                    void* __restrict__ Cout, const float* __restrict__ Resid,
                    int M, int Nc, int Kt, int lda, int ldc,
                    long long strideA, long long strideB, long long strideC,
                    long long strideR)
{
    const int b  = blockIdx.z;
    A  += (long long)b * strideA;
    Bp += (long long)b * strideB;

    const int n0 = blockIdx.x * 128;
    const int m0 = blockIdx.y * 64;
    if (CAUSAL_SKIP && n0 > m0 + 63) return;   // tile fully above diagonal

    const int wave = threadIdx.x >> 5;
    const int lane = threadIdx.x & 31;
    const int half = lane >> 4;
    const int l    = lane & 15;
    const int wm   = m0 + (wave >> 1) * 32;    // wave tile origin (rows), 2x16
    const int wn   = n0 + (wave & 1) * 64;     // wave tile origin (cols), 4x16

    int kbound = Kt;
    if (KLIMIT) kbound = (Kt < n0 + 128) ? Kt : n0 + 128;  // P rows beyond the
                                                           // block's max query are 0
    f32x8 acc[2][4] = {};

    for (int k0 = 0; k0 < kbound; k0 += 32) {
        ABFrag af[2], bf[4];
#pragma unroll
        for (int t = 0; t < 2; ++t) {
            const __bf16* ap = A + (long long)(wm + t * 16 + l) * lda + k0 + half * 16;
            af[t].q[0] = *(const u32x4*)(ap);
            af[t].q[1] = *(const u32x4*)(ap + 8);
        }
        const long long kgrp = (long long)((k0 >> 4) + half) * Nc;
#pragma unroll
        for (int t = 0; t < 4; ++t) {
            const __bf16* bp = Bp + (kgrp + (wn + t * 16 + l)) * 16;
            bf[t].q[0] = *(const u32x4*)(bp);
            bf[t].q[1] = *(const u32x4*)(bp + 8);
        }
#pragma unroll
        for (int i = 0; i < 2; ++i)
#pragma unroll
            for (int j = 0; j < 4; ++j)
                acc[i][j] = __builtin_amdgcn_wmma_f32_16x16x32_bf16(
                    false, af[i].v, false, bf[j].v, (short)0, acc[i][j], false, false);
    }

    // ---- epilogue -----------------------------------------------------------
#pragma unroll
    for (int i = 0; i < 2; ++i) {
#pragma unroll
        for (int j = 0; j < 4; ++j) {
#pragma unroll
            for (int v = 0; v < 8; ++v) {
                const int row = wm + i * 16 + v + (half << 3);
                const int col = wn + j * 16 + l;
                const float val = acc[i][j][v];
                if (EPI == 0) {
                    float* C = (float*)Cout + (long long)b * strideC;
                    C[(long long)row * ldc + col] = val;
                } else if (EPI == 1) {
                    __bf16* C = (__bf16*)Cout + (long long)b * strideC;
                    C[((long long)(row >> 4) * ldc + col) * 16 + (row & 15)] = (__bf16)val;
                } else if (EPI == 2) {
                    __bf16* C = (__bf16*)Cout + (long long)b * strideC;
                    C[(long long)col * ldc + row] = (__bf16)val;   // transposed store
                } else {
                    const float* R = Resid + (long long)b * strideR;
                    float* C = (float*)Cout + (long long)b * strideC;
                    C[(long long)row * ldc + col] = R[(long long)row * ldc + col] + val;
                }
            }
        }
    }
}

// ---------------------------------------------------------------------------
// Casts / layout shuffles
// ---------------------------------------------------------------------------
__global__ void cast_std_bf16(const float* __restrict__ in, __bf16* __restrict__ out,
                              long long n)
{
    long long i = (long long)blockIdx.x * blockDim.x + threadIdx.x;
    if (i < n) out[i] = (__bf16)in[i];
}

// f32 (rows x cols, per batch) -> bf16 K16-grouped: out[((r/16)*cols + c)*16 + r%16]
__global__ void cast_packed_bf16(const float* __restrict__ in, __bf16* __restrict__ out,
                                 int rows, int cols, int nb)
{
    long long total = (long long)nb * rows * cols;
    long long i = (long long)blockIdx.x * blockDim.x + threadIdx.x;
    if (i >= total) return;
    const int c = (int)(i % cols);
    const int r = (int)((i / cols) % rows);
    const int b = (int)(i / ((long long)cols * rows));
    const long long base = (long long)b * rows * cols;
    out[base + ((long long)(r >> 4) * cols + c) * 16 + (r & 15)] = (__bf16)in[i];
}

// W_O (K x D) f32  ->  W_O^T (D x K) bf16
__global__ void transpose_cast_bf16(const float* __restrict__ in, __bf16* __restrict__ out,
                                    int K, int D)
{
    long long i = (long long)blockIdx.x * blockDim.x + threadIdx.x;
    if (i >= (long long)K * D) return;
    const int d = (int)(i % D);
    const int k = (int)(i / D);
    out[(long long)d * K + k] = (__bf16)in[i];
}

// ---------------------------------------------------------------------------
// Causal row softmax: reads f32 logits row i (only j <= i; tiles above the
// diagonal were never written), writes bf16 probabilities TRANSPOSED and
// K16-grouped into Pt so the ctx GEMM can consume it as a B operand:
//   Pt element (t=j, n=i) at [((j/16)*N + i)*16 + (j%16)];  zero for j > i.
// ---------------------------------------------------------------------------
__global__ __launch_bounds__(256)
void softmax_causal(const float* __restrict__ S, __bf16* __restrict__ Pt,
                    int N, long long sStride, long long pStride)
{
    const int i = blockIdx.x;
    const int b = blockIdx.y;
    const float* s = S + (long long)b * sStride + (long long)i * N;
    __bf16* p = Pt + (long long)b * pStride;
    const int len = i + 1;
    const int tid = threadIdx.x;
    __shared__ float red[256];

    float m = -INFINITY;
    for (int j = tid; j < len; j += 256) m = fmaxf(m, s[j]);
    red[tid] = m; __syncthreads();
    for (int off = 128; off > 0; off >>= 1) {
        if (tid < off) red[tid] = fmaxf(red[tid], red[tid + off]);
        __syncthreads();
    }
    m = red[0]; __syncthreads();

    float sum = 0.f;
    for (int j = tid; j < len; j += 256) sum += __expf(s[j] - m);
    red[tid] = sum; __syncthreads();
    for (int off = 128; off > 0; off >>= 1) {
        if (tid < off) red[tid] += red[tid + off];
        __syncthreads();
    }
    const float inv = 1.0f / red[0];

    for (int j = tid; j < N; j += 256) {
        const float pv = (j < len) ? __expf(s[j] - m) * inv : 0.0f;
        p[((long long)(j >> 4) * N + i) * 16 + (j & 15)] = (__bf16)pv;
    }
}

// ---------------------------------------------------------------------------
// Host-side orchestration
// ---------------------------------------------------------------------------
extern "C" void kernel_launch(void* const* d_in, const int* in_sizes, int n_in,
                              void* d_out, int out_size, void* d_ws, size_t ws_size,
                              hipStream_t stream)
{
    (void)in_sizes; (void)n_in; (void)out_size; (void)ws_size;
    const float* X  = (const float*)d_in[0];  // (B, D, N)
    const float* Wq = (const float*)d_in[1];  // (K, D)
    const float* Wk = (const float*)d_in[2];
    const float* Wv = (const float*)d_in[3];
    const float* Wo = (const float*)d_in[4];

    char*  ws  = (char*)d_ws;
    size_t off = 0;
    auto carve = [&](size_t bytes) -> void* {
        void* p = ws + off;
        off = (off + bytes + 255) & ~(size_t)255;
        return p;
    };
    const long long BDN = (long long)CB * CD * CN;
    const long long KD  = (long long)CK * CD;
    const long long BKN = (long long)CB * CK * CN;
    const long long BNN = (long long)CB * CN * CN;

    __bf16* Xstd = (__bf16*)carve(BDN * 2);   // X row-major (A operand of ctx GEMM)
    __bf16* Xp   = (__bf16*)carve(BDN * 2);   // X K16-grouped (B of proj GEMMs)
    __bf16* Wqb  = (__bf16*)carve(KD * 2);
    __bf16* Wkb  = (__bf16*)carve(KD * 2);
    __bf16* Wvb  = (__bf16*)carve(KD * 2);
    __bf16* Wot  = (__bf16*)carve(KD * 2);    // W_O^T, D x K
    __bf16* Qt   = (__bf16*)carve(BKN * 2);   // Q^T, per batch N x K row-major
    __bf16* Kp   = (__bf16*)carve(BKN * 2);   // K proj, K16-grouped K x N
    float*  S    = (float*) carve(BNN * 4);   // logits f32
    __bf16* Pt   = (__bf16*)carve(BNN * 2);   // probs^T K16-grouped N x N
    __bf16* Ctxp = (__bf16*)carve(BDN * 2);   // ctx K16-grouped D x N
    __bf16* VCp  = (__bf16*)carve(BKN * 2);   // W_V*ctx K16-grouped K x N

    // --- precision casts -----------------------------------------------------
    cast_std_bf16<<<dim3((unsigned)((BDN + 255) / 256)), 256, 0, stream>>>(X, Xstd, BDN);
    cast_packed_bf16<<<dim3((unsigned)((BDN + 255) / 256)), 256, 0, stream>>>(X, Xp, CD, CN, CB);
    cast_std_bf16<<<dim3((unsigned)((KD + 255) / 256)), 256, 0, stream>>>(Wq, Wqb, KD);
    cast_std_bf16<<<dim3((unsigned)((KD + 255) / 256)), 256, 0, stream>>>(Wk, Wkb, KD);
    cast_std_bf16<<<dim3((unsigned)((KD + 255) / 256)), 256, 0, stream>>>(Wv, Wvb, KD);
    transpose_cast_bf16<<<dim3((unsigned)((KD + 255) / 256)), 256, 0, stream>>>(Wo, Wot, CK, CD);

    // --- Q^T = (W_Q * X)^T  : M=K, Nc=N, Kt=D -------------------------------
    gemm_bf16_wmma<2, false, false><<<dim3(CN / 128, CK / 64, CB), 128, 0, stream>>>(
        Wqb, Xp, Qt, nullptr, CK, CN, CD, CD, CK,
        0, (long long)CD * CN, (long long)CN * CK, 0);

    // --- Kp = W_K * X (K16-grouped) ------------------------------------------
    gemm_bf16_wmma<1, false, false><<<dim3(CN / 128, CK / 64, CB), 128, 0, stream>>>(
        Wkb, Xp, Kp, nullptr, CK, CN, CD, CD, CN,
        0, (long long)CD * CN, (long long)CK * CN, 0);

    // --- S = Q^T * Kp (f32 logits, skip fully-masked tiles) ------------------
    gemm_bf16_wmma<0, true, false><<<dim3(CN / 128, CN / 64, CB), 128, 0, stream>>>(
        Qt, Kp, S, nullptr, CN, CN, CK, CK, CN,
        (long long)CN * CK, (long long)CK * CN, (long long)CN * CN, 0);

    // --- causal softmax -> Pt (transposed, K16-grouped) ----------------------
    softmax_causal<<<dim3(CN, CB), 256, 0, stream>>>(
        S, Pt, CN, (long long)CN * CN, (long long)CN * CN);

    // --- ctx = X * P^T (K16-grouped out), k-loop clipped by causal zeros -----
    gemm_bf16_wmma<1, false, true><<<dim3(CN / 128, CD / 64, CB), 128, 0, stream>>>(
        Xstd, Pt, Ctxp, nullptr, CD, CN, CN, CN, CN,
        (long long)CD * CN, (long long)CN * CN, (long long)CD * CN, 0);

    // --- VC = W_V * ctx (K16-grouped) ----------------------------------------
    gemm_bf16_wmma<1, false, false><<<dim3(CN / 128, CK / 64, CB), 128, 0, stream>>>(
        Wvb, Ctxp, VCp, nullptr, CK, CN, CD, CD, CN,
        0, (long long)CD * CN, (long long)CK * CN, 0);

    // --- out = X + W_O^T * VC (f32) ------------------------------------------
    gemm_bf16_wmma<3, false, false><<<dim3(CN / 128, CD / 64, CB), 128, 0, stream>>>(
        Wot, VCp, d_out, X, CD, CN, CK, CK, CN,
        0, (long long)CK * CN, (long long)CD * CN, (long long)CD * CN);
}
